// EncoderLayer_38568806318150
// MI455X (gfx1250) — matrix operand order repaired and checked by reading, not compiled, not run on hardware
//
#include <hip/hip_runtime.h>
#include <math.h>

#define Bc 2
#define Sc 2048
#define Dc 1024
#define Hc 4
#define Ic 2048
#define DHc 512
#define FFc 4096
#define BSc (Bc*Sc)

typedef __attribute__((ext_vector_type(16))) __bf16 v16bf;
typedef __attribute__((ext_vector_type(8)))  float  v8f;

union Frag { v16bf v; uint4 u[2]; unsigned short s[16]; };

__device__ __forceinline__ float bf2f(unsigned short u){
  union { float f; unsigned int i; } c; c.i = ((unsigned int)u) << 16; return c.f;
}
__device__ __forceinline__ unsigned short f2bf(float f){
  union { float f; unsigned int i; } c; c.f = f;
  unsigned int r = c.i + 0x7FFFu + ((c.i >> 16) & 1u);
  return (unsigned short)(r >> 16);
}
__device__ __forceinline__ v8f v8zero(){
  v8f z = {0.f,0.f,0.f,0.f,0.f,0.f,0.f,0.f}; return z;
}
__device__ __forceinline__ v8f wmma_bf16(const Frag& a, const Frag& b, v8f c){
  return __builtin_amdgcn_wmma_f32_16x16x32_bf16(false, a.v, false, b.v, (short)0, c, false, false);
}

// ---- CDNA5 async global->LDS copy (ASYNCcnt path), 16B per lane ----
// Low 32 bits of a generic shared-aperture address are the LDS byte offset
// so we can pass pointer bits directly.
__device__ __forceinline__ void async_g2l_b128(const void* gsrc, void* lds_dst){
#if defined(__gfx1250__)
  unsigned lds = (unsigned)(unsigned long long)(uintptr_t)lds_dst;
  unsigned long long ga = (unsigned long long)(uintptr_t)gsrc;
  asm volatile("global_load_async_to_lds_b128 %0, %1, off"
               :: "v"(lds), "v"(ga) : "memory");
#else
  *(uint4*)lds_dst = *(const uint4*)gsrc;
#endif
}
__device__ __forceinline__ void async_wait(){
#if defined(__gfx1250__)
  asm volatile("s_wait_asynccnt 0x0" ::: "memory");
#endif
}

// ---------------- fp32 -> bf16 weight conversion ----------------
__global__ __launch_bounds__(256) void k_f32_to_bf16(const float* __restrict__ src,
                                                     unsigned short* __restrict__ dst, int n){
  int i = blockIdx.x*256 + threadIdx.x;
  if (i < n) dst[i] = f2bf(src[i]);
}

// ---------------- LayerNorm over D=1024, writes bf16 ----------------
__global__ __launch_bounds__(256) void k_layernorm(const float* __restrict__ x,
                                                   const float* __restrict__ g,
                                                   const float* __restrict__ bta,
                                                   unsigned short* __restrict__ xn){
  __shared__ float r1[256], r2[256];
  int row = blockIdx.x, tid = threadIdx.x;
  const float* xr = x + (size_t)row * Dc;
  float v[4]; float s = 0.f, ss = 0.f;
  #pragma unroll
  for (int j=0;j<4;j++){ v[j] = xr[tid + j*256]; s += v[j]; ss += v[j]*v[j]; }
  r1[tid]=s; r2[tid]=ss; __syncthreads();
  for (int o=128;o>0;o>>=1){ if (tid<o){ r1[tid]+=r1[tid+o]; r2[tid]+=r2[tid+o]; } __syncthreads(); }
  float mu = r1[0]/(float)Dc;
  float var = r2[0]/(float)Dc - mu*mu;
  float inv = rsqrtf(var + 1e-6f);
  unsigned short* o = xn + (size_t)row*Dc;
  #pragma unroll
  for (int j=0;j<4;j++){ int c = tid + j*256; o[c] = f2bf((v[j]-mu)*inv*g[c] + bta[c]); }
}

// ---------------- generic bf16 WMMA GEMM, 128x128 block, K-step 64 ----------------
#define ASTRD 72
#define BSTRD 72

template<int MODE> // 0: bias->bf16, 1: bias+relu->bf16, 2: down (x2 f32 + DyT bf16), 3: final (bias + residual -> f32)
__global__ __launch_bounds__(256) void k_gemm(
  const unsigned short* __restrict__ A, int lda,
  const unsigned short* __restrict__ Bw,       // [K][N] row-major bf16
  const float* __restrict__ bias,
  int N, int K,
  unsigned short* __restrict__ obf, float* __restrict__ of32,
  const float* __restrict__ xres, const float* __restrict__ dytA,
  const float* __restrict__ dytW, const float* __restrict__ dytB)
{
  __shared__ __align__(16) unsigned short sA[128*ASTRD];
  __shared__ __align__(16) unsigned short sB[128*BSTRD];  // transposed [n][k]
  int tid = threadIdx.x;
  int wid = tid>>5, lane = tid&31, lh = lane>>4, l15 = lane&15;
  int m0 = blockIdx.y*128, n0 = blockIdx.x*128;
  int wm = (wid>>1)*32, wn = (wid&1)*64;
  v8f acc[2][4];
  #pragma unroll
  for (int i=0;i<2;i++)
    #pragma unroll
    for (int j=0;j<4;j++) acc[i][j] = v8zero();

  int ar = tid>>1, ak = (tid&1)*32;           // A: 128 rows x 64 k, 32 elems/thread
  int br = tid>>2, bc = (tid&3)*32;           // B: 64 k x 128 n, 32 elems/thread
  const unsigned short* aSrc = A  + (size_t)(m0+ar)*lda + ak;
  const unsigned short* bSrc = Bw + (size_t)br*N + n0 + bc;
  unsigned short* aDst = sA + ar*ASTRD + ak;

  for (int k0=0; k0<K; k0+=64){
    if (k0 + 64 < K) __builtin_prefetch(bSrc + (size_t)(k0+64)*N, 0, 1);
    // A tile: async DMA straight into LDS (no VGPR round trip)
    #pragma unroll
    for (int e=0;e<4;e++) async_g2l_b128(aSrc + k0 + e*8, aDst + e*8);
    // B tile: VGPR path (needs 2-byte transpose scatter)
    Frag b0, b1;
    b0.u[0] = *(const uint4*)(bSrc + (size_t)k0*N);
    b0.u[1] = *(const uint4*)(bSrc + (size_t)k0*N + 8);
    b1.u[0] = *(const uint4*)(bSrc + (size_t)k0*N + 16);
    b1.u[1] = *(const uint4*)(bSrc + (size_t)k0*N + 24);
    #pragma unroll
    for (int e=0;e<16;e++) sB[(bc+e)*BSTRD + br] = b0.s[e];
    #pragma unroll
    for (int e=0;e<16;e++) sB[(bc+16+e)*BSTRD + br] = b1.s[e];
    async_wait();
    __syncthreads();

    #pragma unroll
    for (int kk=0; kk<64; kk+=32){
      Frag af[2], bfm[4];
      #pragma unroll
      for (int tm=0;tm<2;tm++){
        int r = wm + tm*16 + l15;
        af[tm].u[0] = *(const uint4*)(sA + r*ASTRD + kk + lh*8);
        af[tm].u[1] = *(const uint4*)(sA + r*ASTRD + kk + 16 + lh*8);
      }
      #pragma unroll
      for (int tn=0;tn<4;tn++){
        int c = wn + tn*16 + l15;
        bfm[tn].u[0] = *(const uint4*)(sB + c*BSTRD + kk + lh*8);
        bfm[tn].u[1] = *(const uint4*)(sB + c*BSTRD + kk + 16 + lh*8);
      }
      #pragma unroll
      for (int tm=0;tm<2;tm++)
        #pragma unroll
        for (int tn=0;tn<4;tn++)
          acc[tm][tn] = wmma_bf16(af[tm], bfm[tn], acc[tm][tn]);
    }
    __syncthreads();
  }

  float da = (MODE==2) ? dytA[0] : 0.0f;
  #pragma unroll
  for (int tm=0;tm<2;tm++){
    #pragma unroll
    for (int tn=0;tn<4;tn++){
      #pragma unroll
      for (int r=0;r<8;r++){
        int m = m0 + wm + tm*16 + r + lh*8;
        int n = n0 + wn + tn*16 + l15;
        size_t idx = (size_t)m*N + n;
        float val = acc[tm][tn][r] + bias[n];
        if (MODE==0){ obf[idx] = f2bf(val); }
        else if (MODE==1){ obf[idx] = f2bf(fmaxf(val, 0.f)); }
        else if (MODE==2){
          float x2 = val + 2.0f*xres[idx];
          of32[idx] = x2;
          obf[idx] = f2bf(dytW[n]*tanhf(da*x2) + dytB[n]);
        } else {
          of32[idx] = val + xres[idx];
        }
      }
    }
  }
}

// ---------------- depthwise causal conv (K=4) + SiLU ----------------
__global__ __launch_bounds__(256) void k_conv_silu(const unsigned short* __restrict__ up,
                                                   const float* __restrict__ cw,
                                                   const float* __restrict__ cb,
                                                   unsigned short* __restrict__ xc){
  int idx = blockIdx.x*256 + threadIdx.x;     // BS*I
  int c  = idx & (Ic-1);
  int bs = idx >> 11;
  int s  = bs & (Sc-1);
  int b  = bs >> 11;
  float acc = cb[c];
  #pragma unroll
  for (int t=0;t<4;t++){
    int sp = s - 3 + t;
    if (sp >= 0) acc += bf2f(up[(size_t)(b*Sc+sp)*(2*Ic) + c]) * cw[t*Ic + c];
  }
  float sig = 1.f/(1.f + __expf(-acc));
  xc[(size_t)bs*Ic + c] = f2bf(acc*sig);
}

// ---------------- gate: g = xc @ W_if + b_if ----------------
__global__ __launch_bounds__(256) void k_gate(const unsigned short* __restrict__ xc,
                                              const float* __restrict__ Wif,
                                              const float* __restrict__ bif,
                                              float* __restrict__ g){
  __shared__ float red[256];
  int bs = blockIdx.x, tid = threadIdx.x;
  float p[8] = {0,0,0,0,0,0,0,0};
  const unsigned short* xr = xc + (size_t)bs*Ic;
  for (int k=tid; k<Ic; k+=256){
    float xv = bf2f(xr[k]);
    const float* wr = Wif + (size_t)k*8;
    #pragma unroll
    for (int c=0;c<8;c++) p[c] += xv * wr[c];
  }
  #pragma unroll
  for (int c=0;c<8;c++){
    red[tid] = p[c]; __syncthreads();
    for (int o=128;o>0;o>>=1){ if (tid<o) red[tid]+=red[tid+o]; __syncthreads(); }
    if (tid==0) g[(size_t)bs*8 + c] = red[0] + bif[c];
    __syncthreads();
  }
}

// ---------------- per-(b,h) scans ----------------
__global__ __launch_bounds__(256) void k_scan(const float* __restrict__ g,
                                              float* __restrict__ aout,
                                              float* __restrict__ cmaxout,
                                              float* __restrict__ mout){
  __shared__ float sc[256];
  int bh = blockIdx.x; int b = bh / Hc; int hh = bh % Hc;
  int tid = threadIdx.x;
  float F[8], av[8];
  float run = 0.f;
  #pragma unroll
  for (int j=0;j<8;j++){
    int s = tid*8 + j;
    float f = g[(size_t)(b*Sc + s)*8 + Hc + hh];
    float ls = fminf(f, 0.f) - log1pf(__expf(-fabsf(f)));
    run += ls; F[j] = run;
  }
  sc[tid] = run; __syncthreads();
  for (int o=1;o<256;o<<=1){
    float v = 0.f; if (tid>=o) v = sc[tid-o];
    __syncthreads();
    sc[tid] += v; __syncthreads();
  }
  float pre = sc[tid] - run;
  float rmax = -__builtin_inff();
  #pragma unroll
  for (int j=0;j<8;j++){
    int s = tid*8 + j;
    F[j] += pre;
    float ip = g[(size_t)(b*Sc + s)*8 + hh];
    av[j] = ip - F[j];
    rmax = fmaxf(rmax, av[j]);
    aout[(size_t)bh*Sc + s] = av[j];
  }
  __syncthreads();
  sc[tid] = rmax; __syncthreads();
  for (int o=1;o<256;o<<=1){
    float v = -__builtin_inff(); if (tid>=o) v = sc[tid-o];
    __syncthreads();
    sc[tid] = fmaxf(sc[tid], v); __syncthreads();
  }
  float cm = (tid>0) ? sc[tid-1] : -__builtin_inff();
  #pragma unroll
  for (int j=0;j<8;j++){
    int s = tid*8 + j;
    cm = fmaxf(cm, av[j]);
    cmaxout[(size_t)bh*Sc + s] = cm;
    mout[(size_t)bh*Sc + s] = F[j] + cm;
  }
}

// ---------------- flash-style mLSTM attention, all WMMA ----------------
#define KSTRD 520
#define VSTRD 40
#define SSTRD 40

__global__ __launch_bounds__(256) void k_attn(
  const unsigned short* __restrict__ qb,
  const unsigned short* __restrict__ kb,
  const unsigned short* __restrict__ vb,
  const float* __restrict__ aarr,
  const float* __restrict__ cmarr,
  const float* __restrict__ marr,
  float* __restrict__ hout)
{
  __shared__ __align__(16) unsigned short sK[32*KSTRD];    // k tile [32 j][512 dh]
  __shared__ __align__(16) unsigned short sV[256*VSTRD];   // vT chunk [256 dh][32 j]
  __shared__ __align__(16) unsigned short sS[64*SSTRD];    // scores [64 i][32 j] bf16
  __shared__ float sSum[64], sCmax[64], sM[64], sAcol[32];

  const int nrb = Sc/64;
  int rb = blockIdx.x % nrb;
  int bh = blockIdx.x / nrb;
  int b = bh / Hc, hh = bh % Hc;
  int i0 = rb*64;
  int tid = threadIdx.x, wid = tid>>5, lane = tid&31, lh = lane>>4, l15 = lane&15;
  const float scale = 0.044194173824159216f; // 1/sqrt(512)

  if (tid < 64){
    sSum[tid]  = 0.f;
    sCmax[tid] = cmarr[(size_t)bh*Sc + i0 + tid];
    sM[tid]    = marr[(size_t)bh*Sc + i0 + tid];
  }

  int tr = wid & 3, tc = wid >> 2;
  Frag qf[16];
  {
    const unsigned short* qrow = qb + (size_t)(b*Sc + i0 + tr*16 + l15)*Ic + hh*DHc;
    #pragma unroll
    for (int ks=0; ks<16; ks++){
      qf[ks].u[0] = *(const uint4*)(qrow + 32*ks + lh*8);
      qf[ks].u[1] = *(const uint4*)(qrow + 32*ks + 16 + lh*8);
    }
  }

  v8f accH[4][4];
  #pragma unroll
  for (int i=0;i<4;i++)
    #pragma unroll
    for (int j=0;j<4;j++) accH[i][j] = v8zero();

  int nblk = (i0>>5) + 2;
  for (int blk=0; blk<nblk; blk++){
    int j0 = blk*32;
    __syncthreads();
    {
      int jl = tid>>3, db = (tid&7)*64;
      const unsigned short* src = kb + (size_t)(b*Sc + j0 + jl)*Ic + hh*DHc + db;
      unsigned short* dst = sK + jl*KSTRD + db;
      #pragma unroll
      for (int e=0;e<8;e++) async_g2l_b128(src + e*8, dst + e*8);
      if (tid < 32) sAcol[tid] = aarr[(size_t)bh*Sc + j0 + tid];
      async_wait();
    }
    __syncthreads();

    // phase 1: P = q @ k^T
    v8f p = v8zero();
    #pragma unroll
    for (int ks=0; ks<16; ks++){
      Frag bf;
      const unsigned short* kp = sK + (tc*16 + l15)*KSTRD + 32*ks + lh*8;
      bf.u[0] = *(const uint4*)(kp);
      bf.u[1] = *(const uint4*)(kp + 16);
      p = wmma_bf16(qf[ks], bf, p);
    }
    #pragma unroll
    for (int r=0;r<8;r++){
      int il = tr*16 + r + lh*8;
      int jl = tc*16 + l15;
      int i = i0 + il, j = j0 + jl;
      float w = (j <= i) ? __expf(sAcol[jl] - sCmax[il]) : 0.f;
      sS[il*SSTRD + jl] = f2bf(p[r]*scale*w);
    }
    __syncthreads();
    if (tid < 64){
      float s = 0.f;
      #pragma unroll
      for (int c=0;c<32;c++) s += bf2f(sS[tid*SSTRD + c]);
      sSum[tid] += s;
    }

    // phase 2: h += P @ v
    #pragma unroll
    for (int pass=0; pass<2; pass++){
      __syncthreads();
      {
        int jl = tid>>3, db = (tid&7)*32;
        const uint4* src = (const uint4*)(vb + (size_t)(b*Sc + j0 + jl)*Ic + hh*DHc + pass*256 + db);
        Frag t0, t1;
        t0.u[0] = src[0]; t0.u[1] = src[1];
        t1.u[0] = src[2]; t1.u[1] = src[3];
        #pragma unroll
        for (int e=0;e<16;e++) sV[(db+e)*VSTRD + jl] = t0.s[e];
        #pragma unroll
        for (int e=0;e<16;e++) sV[(db+16+e)*VSTRD + jl] = t1.s[e];
      }
      __syncthreads();
      Frag bf0, bf1;
      {
        const unsigned short* vp0 = sV + (wid*32 + l15)*VSTRD + lh*8;
        bf0.u[0] = *(const uint4*)(vp0); bf0.u[1] = *(const uint4*)(vp0 + 16);
        const unsigned short* vp1 = sV + (wid*32 + 16 + l15)*VSTRD + lh*8;
        bf1.u[0] = *(const uint4*)(vp1); bf1.u[1] = *(const uint4*)(vp1 + 16);
      }
      #pragma unroll
      for (int tm=0;tm<4;tm++){
        Frag af;
        const unsigned short* sp = sS + (tm*16 + l15)*SSTRD;
        af.u[0] = *(const uint4*)(sp + lh*8);
        af.u[1] = *(const uint4*)(sp + 16 + lh*8);
        accH[tm][pass*2+0] = wmma_bf16(af, bf0, accH[tm][pass*2+0]);
        accH[tm][pass*2+1] = wmma_bf16(af, bf1, accH[tm][pass*2+1]);
      }
    }
  }
  __syncthreads();
  #pragma unroll
  for (int tm=0;tm<4;tm++){
    #pragma unroll
    for (int q=0;q<4;q++){
      int pass = q>>1, tn = q&1;
      #pragma unroll
      for (int r=0;r<8;r++){
        int il = tm*16 + r + lh*8;
        int i = i0 + il;
        int dh = pass*256 + wid*32 + tn*16 + l15;
        float n = fmaxf(fabsf(sSum[il]), __expf(-sM[il]));
        hout[((size_t)bh*Sc + i)*DHc + dh] = accH[tm][q][r] / n;
      }
    }
  }
}

// ---------------- per-(b,h,s) head LayerNorm * mh_g * silu(z) ----------------
__global__ __launch_bounds__(256) void k_headnorm(const float* __restrict__ h,
                                                  const float* __restrict__ mhg,
                                                  const unsigned short* __restrict__ up,
                                                  unsigned short* __restrict__ hob){
  __shared__ float r1[256], r2[256];
  int idx = blockIdx.x;
  int s  = idx & (Sc-1);
  int bh = idx >> 11;
  int b = bh / Hc, hh = bh % Hc;
  int tid = threadIdx.x;
  const float* hr = h + ((size_t)bh*Sc + s)*DHc;
  float v0 = hr[tid], v1 = hr[tid+256];
  r1[tid] = v0+v1; r2[tid] = v0*v0+v1*v1; __syncthreads();
  for (int o=128;o>0;o>>=1){ if (tid<o){ r1[tid]+=r1[tid+o]; r2[tid]+=r2[tid+o]; } __syncthreads(); }
  float mu = r1[0]/(float)DHc, var = r2[0]/(float)DHc - mu*mu;
  float inv = rsqrtf(var + 1e-6f);
  size_t row = (size_t)(b*Sc + s);
  #pragma unroll
  for (int j=0;j<2;j++){
    int d = tid + j*256;
    int col = hh*DHc + d;
    float hn = ((j ? v1 : v0) - mu)*inv;
    float z = bf2f(up[row*(2*Ic) + Ic + col]);
    float sz = z / (1.f + __expf(-z));
    hob[row*Ic + col] = f2bf(hn * mhg[col] * sz);
  }
}

// ----------------------------------------------------------------------------
extern "C" void kernel_launch(void* const* d_in, const int* in_sizes, int n_in,
                              void* d_out, int out_size, void* d_ws, size_t ws_size,
                              hipStream_t stream) {
  (void)in_sizes; (void)n_in; (void)out_size; (void)ws_size;
  const float* x      = (const float*)d_in[0];
  const float* ln_g   = (const float*)d_in[1];
  const float* ln_b   = (const float*)d_in[2];
  const float* W_up   = (const float*)d_in[3];
  const float* b_up   = (const float*)d_in[4];
  const float* conv_w = (const float*)d_in[5];
  const float* conv_b = (const float*)d_in[6];
  const float* Wq     = (const float*)d_in[7];
  const float* bq     = (const float*)d_in[8];
  const float* Wk     = (const float*)d_in[9];
  const float* bk     = (const float*)d_in[10];
  const float* Wv     = (const float*)d_in[11];
  const float* bv     = (const float*)d_in[12];
  const float* W_if   = (const float*)d_in[13];
  const float* b_if   = (const float*)d_in[14];
  const float* mh_g   = (const float*)d_in[15];
  const float* W_down = (const float*)d_in[16];
  const float* b_down = (const float*)d_in[17];
  const float* dyt_a  = (const float*)d_in[18];
  const float* dyt_w  = (const float*)d_in[19];
  const float* dyt_b  = (const float*)d_in[20];
  const float* W1     = (const float*)d_in[21];
  const float* b1     = (const float*)d_in[22];
  const float* W2     = (const float*)d_in[23];
  const float* b2     = (const float*)d_in[24];

  char* ws = (char*)d_ws;
  size_t off = 0;
  auto alloc = [&](size_t bytes)->char* {
    char* p = ws + off; off += (bytes + 255) & ~(size_t)255; return p;
  };
  unsigned short* xn    = (unsigned short*)alloc((size_t)BSc*Dc*2);
  unsigned short* wupb  = (unsigned short*)alloc((size_t)Dc*(2*Ic)*2);
  unsigned short* upb   = (unsigned short*)alloc((size_t)BSc*(2*Ic)*2);
  unsigned short* xcb   = (unsigned short*)alloc((size_t)BSc*Ic*2);
  unsigned short* wqb   = (unsigned short*)alloc((size_t)Ic*Ic*2);
  unsigned short* wkb   = (unsigned short*)alloc((size_t)Ic*Ic*2);
  unsigned short* wvb   = (unsigned short*)alloc((size_t)Ic*Ic*2);
  unsigned short* qbuf  = (unsigned short*)alloc((size_t)BSc*Ic*2);
  unsigned short* kbuf  = (unsigned short*)alloc((size_t)BSc*Ic*2);
  unsigned short* vbuf  = (unsigned short*)alloc((size_t)BSc*Ic*2);
  float*          gbuf  = (float*)alloc((size_t)BSc*8*4);
  float*          abuf  = (float*)alloc((size_t)Bc*Hc*Sc*4);
  float*          cmbuf = (float*)alloc((size_t)Bc*Hc*Sc*4);
  float*          mbuf  = (float*)alloc((size_t)Bc*Hc*Sc*4);
  float*          hbuf  = (float*)alloc((size_t)Bc*Hc*Sc*DHc*4);
  unsigned short* hob   = (unsigned short*)alloc((size_t)BSc*Ic*2);
  unsigned short* wdb   = (unsigned short*)alloc((size_t)Ic*Dc*2);
  float*          x2buf = (float*)alloc((size_t)BSc*Dc*4);
  unsigned short* tbuf  = (unsigned short*)alloc((size_t)BSc*Dc*2);
  unsigned short* w1b   = (unsigned short*)alloc((size_t)Dc*FFc*2);
  unsigned short* w2b   = (unsigned short*)alloc((size_t)FFc*Dc*2);
  unsigned short* f1b   = (unsigned short*)alloc((size_t)BSc*FFc*2);

  const int T = 256;
  auto cvt = [&](const float* s, unsigned short* d, int n){
    k_f32_to_bf16<<<(n+T-1)/T, T, 0, stream>>>(s, d, n);
  };
  cvt(W_up,   wupb, Dc*2*Ic);
  cvt(Wq,     wqb,  Ic*Ic);
  cvt(Wk,     wkb,  Ic*Ic);
  cvt(Wv,     wvb,  Ic*Ic);
  cvt(W_down, wdb,  Ic*Dc);
  cvt(W1,     w1b,  Dc*FFc);
  cvt(W2,     w2b,  FFc*Dc);

  k_layernorm<<<BSc, T, 0, stream>>>(x, ln_g, ln_b, xn);

  k_gemm<0><<<dim3((2*Ic)/128, BSc/128), T, 0, stream>>>(xn, Dc, wupb, b_up, 2*Ic, Dc,
                                                         upb, nullptr, nullptr, nullptr, nullptr, nullptr);
  k_conv_silu<<<(BSc*Ic)/T, T, 0, stream>>>(upb, conv_w, conv_b, xcb);

  k_gemm<0><<<dim3(Ic/128, BSc/128), T, 0, stream>>>(xcb, Ic,   wqb, bq, Ic, Ic,
                                                     qbuf, nullptr, nullptr, nullptr, nullptr, nullptr);
  k_gemm<0><<<dim3(Ic/128, BSc/128), T, 0, stream>>>(xcb, Ic,   wkb, bk, Ic, Ic,
                                                     kbuf, nullptr, nullptr, nullptr, nullptr, nullptr);
  k_gemm<0><<<dim3(Ic/128, BSc/128), T, 0, stream>>>(upb, 2*Ic, wvb, bv, Ic, Ic,
                                                     vbuf, nullptr, nullptr, nullptr, nullptr, nullptr);

  k_gate<<<BSc, T, 0, stream>>>(xcb, W_if, b_if, gbuf);
  k_scan<<<Bc*Hc, T, 0, stream>>>(gbuf, abuf, cmbuf, mbuf);

  k_attn<<<Bc*Hc*(Sc/64), T, 0, stream>>>(qbuf, kbuf, vbuf, abuf, cmbuf, mbuf, hbuf);

  k_headnorm<<<Bc*Hc*Sc, T, 0, stream>>>(hbuf, mh_g, upb, hob);

  k_gemm<2><<<dim3(Dc/128, BSc/128), T, 0, stream>>>(hob, Ic, wdb, b_down, Dc, Ic,
                                                     tbuf, x2buf, x, dyt_a, dyt_w, dyt_b);
  k_gemm<1><<<dim3(FFc/128, BSc/128), T, 0, stream>>>(tbuf, Dc, w1b, b1, FFc, Dc,
                                                      f1b, nullptr, nullptr, nullptr, nullptr, nullptr);
  k_gemm<3><<<dim3(Dc/128, BSc/128), T, 0, stream>>>(f1b, FFc, w2b, b2, Dc, FFc,
                                                     nullptr, (float*)d_out, x2buf, nullptr, nullptr, nullptr);
}
